// Codebook_52192442581124
// MI455X (gfx1250) — matrix operand order repaired and checked by reading, not compiled, not run on hardware
//
#include <hip/hip_runtime.h>
#include <hip/hip_bf16.h>

typedef __attribute__((ext_vector_type(16))) __bf16 v16bf;
typedef __attribute__((ext_vector_type(8)))  float  v8f;

#define B_ROWS   4096
#define N_CODES  16384
#define KDIM     1024
#define TOPK     32

#define ROWS     64                   // x rows per workgroup panel
#define SLICES   16                   // N-dimension split for grid fill
#define SLICE_N  (N_CODES / SLICES)   // 1024 codebook rows per WG
#define NTILE    128                  // codebook cols per inner tile (4 col-waves * 32)
#define A_STRIDE 1032                 // bf16 elems; 2064B stride breaks bank conflicts
#define S_STRIDE 133                  // f32 elems; 133 mod 64 = 5 -> conflict-free scans

#define LDS_A_BYTES  (ROWS * A_STRIDE * 2)               // 132096
#define LDS_S_BYTES  (ROWS * S_STRIDE * 4)               // 34048
#define LDS_QV_BYTES (256 * TOPK * 4)                    // 32768
#define LDS_QI_BYTES (256 * TOPK * 4)                    // 32768
#define SMEM_BYTES   (LDS_A_BYTES + LDS_S_BYTES + LDS_QV_BYTES + LDS_QI_BYTES)

#define NEG_INF (-3.4e38f)

__device__ __forceinline__ unsigned short f2bf(float f) {
    unsigned int u = __builtin_bit_cast(unsigned int, f);
    u += 0x7FFFu + ((u >> 16) & 1u);      // round to nearest even
    return (unsigned short)(u >> 16);
}

// ---------------------------------------------------------------------------
// row-normalize fp32 [rows x 1024] -> bf16, clamping norm at 1e-8
// ---------------------------------------------------------------------------
__global__ __launch_bounds__(256) void normalize_bf16(
    const float* __restrict__ src, unsigned short* __restrict__ dst)
{
    __shared__ float red[256];
    const int row = blockIdx.x;
    const int tid = threadIdx.x;
    const float4 v = *(const float4*)(src + (size_t)row * KDIM + tid * 4);
    red[tid] = v.x * v.x + v.y * v.y + v.z * v.z + v.w * v.w;
    __syncthreads();
    for (int s = 128; s > 0; s >>= 1) {
        if (tid < s) red[tid] += red[tid + s];
        __syncthreads();
    }
    const float scale = 1.0f / fmaxf(sqrtf(red[0]), 1e-8f);
    const unsigned short b0 = f2bf(v.x * scale), b1 = f2bf(v.y * scale);
    const unsigned short b2 = f2bf(v.z * scale), b3 = f2bf(v.w * scale);
    uint2 o;
    o.x = (unsigned)b0 | ((unsigned)b1 << 16);
    o.y = (unsigned)b2 | ((unsigned)b3 << 16);
    *(uint2*)(dst + (size_t)row * KDIM + tid * 4) = o;
}

// ---------------------------------------------------------------------------
// fused bf16 WMMA cosine-sim GEMM + per-(row,slice) partial top-32
// grid = (SLICES, 4096/ROWS) = (16, 64) -> 1024 WGs; 256 threads = 8 waves.
// Wave (wr, wc) with wr = wave>>2, wc = wave&3 owns a 32x32 output block
// (2 row tiles x 2 col tiles -> 4 accumulators, B frags reused x2, A frags x2).
// ---------------------------------------------------------------------------
__global__ __launch_bounds__(256) void fused_cos_topk(
    const unsigned short* __restrict__ xn,   // [4096 x 1024] bf16 normalized x
    const unsigned short* __restrict__ cn,   // [16384 x 1024] bf16 normalized codebook
    float* __restrict__ pvals,               // [4096 x SLICES x 32] partial values
    int*   __restrict__ pidx)                // [4096 x SLICES x 32] partial indices
{
    extern __shared__ char smem[];
    unsigned short* lds_a  = (unsigned short*)smem;
    float*          scores = (float*)(smem + LDS_A_BYTES);
    float*          qvals  = (float*)(smem + LDS_A_BYTES + LDS_S_BYTES);
    int*            qidx   = (int*)  (smem + LDS_A_BYTES + LDS_S_BYTES + LDS_QV_BYTES);

    const int tid  = threadIdx.x;
    const int wave = tid >> 5;
    const int lane = tid & 31;
    const int half = lane >> 4;            // 0: lanes 0-15, 1: lanes 16-31
    const int l15  = lane & 15;
    const int wr   = wave >> 2;            // row-group 0..1  (32 rows each)
    const int wc   = wave & 3;             // col-group 0..3  (32 cols each)

    const int rowbase   = blockIdx.y * ROWS;
    const int slicebase = blockIdx.x * SLICE_N;

    // --- stage the 64 x 1024 bf16 A-panel into LDS (16B chunks) ---
    for (int chunk = tid; chunk < ROWS * (KDIM / 8); chunk += 256) {
        const int r  = chunk >> 7;
        const int c8 = (chunk & 127) << 3;
        *(uint4*)&lds_a[r * A_STRIDE + c8] =
            *(const uint4*)&xn[(size_t)(rowbase + r) * KDIM + c8];
    }
    // --- every thread owns one private top-32 list for (row = tid>>2, chunk = tid&3)
    const int srow   = tid >> 2;
    const int schunk = tid & 3;
    const int q      = tid * TOPK;
    for (int j = 0; j < TOPK; ++j) { qvals[q + j] = NEG_INF; qidx[q + j] = 0; }
    __syncthreads();

    float minv = NEG_INF;
    int   minpos = 0;

    // A fragment bases (16-bit A 16x32 layout: lane l15/half -> K half*8, 16+half*8)
    const unsigned short* aptr0 = &lds_a[(wr * 32 + l15) * A_STRIDE + half * 8];
    const unsigned short* aptr1 = aptr0 + 16 * A_STRIDE;

    for (int nb = 0; nb < SLICE_N; nb += NTILE) {
        // B fragments (16-bit B 32x16 layout: lane = col, lane half -> K half*8, 16+half*8)
        const size_t col0 = (size_t)(slicebase + nb) + wc * 32 + l15;
        const unsigned short* bptr0 = cn + col0 * KDIM + half * 8;
        const unsigned short* bptr1 = bptr0 + (size_t)16 * KDIM;

        v8f a00 = {0.f,0.f,0.f,0.f,0.f,0.f,0.f,0.f};
        v8f a10 = a00, a01 = a00, a11 = a00;

        for (int kk = 0; kk < KDIM; kk += 32) {
            union { v16bf v; uint4 u[2]; } fa0, fa1, fb0, fb1;
            fb0.u[0] = *(const uint4*)(bptr0 + kk);
            fb0.u[1] = *(const uint4*)(bptr0 + kk + 16);
            fb1.u[0] = *(const uint4*)(bptr1 + kk);
            fb1.u[1] = *(const uint4*)(bptr1 + kk + 16);
            fa0.u[0] = *(const uint4*)(aptr0 + kk);
            fa0.u[1] = *(const uint4*)(aptr0 + kk + 16);
            fa1.u[0] = *(const uint4*)(aptr1 + kk);
            fa1.u[1] = *(const uint4*)(aptr1 + kk + 16);
            a00 = __builtin_amdgcn_wmma_f32_16x16x32_bf16(false, fa0.v, false, fb0.v, (short)0, a00, false, false);
            a10 = __builtin_amdgcn_wmma_f32_16x16x32_bf16(false, fa1.v, false, fb0.v, (short)0, a10, false, false);
            a01 = __builtin_amdgcn_wmma_f32_16x16x32_bf16(false, fa0.v, false, fb1.v, (short)0, a01, false, false);
            a11 = __builtin_amdgcn_wmma_f32_16x16x32_bf16(false, fa1.v, false, fb1.v, (short)0, a11, false, false);
        }

        // C layout: row = vgpr + 8*half
        #pragma unroll
        for (int v = 0; v < 8; ++v) {
            const int r0 = wr * 32 + half * 8 + v;
            const int cb = wc * 32 + l15;
            scores[r0 * S_STRIDE + cb]             = a00[v];
            scores[(r0 + 16) * S_STRIDE + cb]      = a10[v];
            scores[r0 * S_STRIDE + cb + 16]        = a01[v];
            scores[(r0 + 16) * S_STRIDE + cb + 16] = a11[v];
        }
        __syncthreads();

        // all 256 threads scan: thread owns 32 cols of one row
        {
            const float* sp = &scores[srow * S_STRIDE + schunk * 32];
            const int nbase = slicebase + nb + schunk * 32;
            for (int s = 0; s < TOPK; ++s) {
                const float v = sp[s];
                if (v > minv) {
                    qvals[q + minpos] = v;
                    qidx[q + minpos]  = nbase + s;
                    minv = qvals[q]; minpos = 0;
                    #pragma unroll
                    for (int j = 1; j < TOPK; ++j) {
                        const float tv = qvals[q + j];
                        if (tv < minv) { minv = tv; minpos = j; }
                    }
                }
            }
        }
        __syncthreads();
    }

    // merge each row's 4 chunk-lists (128 cands) -> top-32 for this slice
    if (tid < ROWS) {
        float* fv = scores;                       // reuse score buffer
        int*   fi = (int*)(scores + ROWS * TOPK);
        const int fb = tid * TOPK;
        const int lb = tid * 4 * TOPK;            // chunk list 0 of this row
        float mv = 3.4e38f; int mp = 0;
        for (int j = 0; j < TOPK; ++j) {
            const float v = qvals[lb + j];
            fv[fb + j] = v; fi[fb + j] = qidx[lb + j];
            if (v < mv) { mv = v; mp = j; }
        }
        for (int c = 1; c < 4; ++c) {
            const int cb = lb + c * TOPK;
            for (int j = 0; j < TOPK; ++j) {
                const float v = qvals[cb + j];
                if (v > mv) {
                    fv[fb + mp] = v; fi[fb + mp] = qidx[cb + j];
                    mv = fv[fb]; mp = 0;
                    for (int t = 1; t < TOPK; ++t) {
                        const float tv = fv[fb + t];
                        if (tv < mv) { mv = tv; mp = t; }
                    }
                }
            }
        }
        const size_t ob = ((size_t)(rowbase + tid) * SLICES + blockIdx.x) * TOPK;
        for (int j = 0; j < TOPK; ++j) { pvals[ob + j] = fv[fb + j]; pidx[ob + j] = fi[fb + j]; }
    }
}

// ---------------------------------------------------------------------------
// merge SLICES partial top-32 lists per row -> final top-32 indices
// one thread per row; private lists in LDS (dynamic indexing)
// ---------------------------------------------------------------------------
__global__ __launch_bounds__(128) void merge_topk(
    const float* __restrict__ pvals, const int* __restrict__ pidx,
    int* __restrict__ topk_out)
{
    __shared__ float lv[128 * TOPK];
    __shared__ int   li[128 * TOPK];
    const int row = blockIdx.x * 128 + threadIdx.x;
    const int q   = threadIdx.x * TOPK;
    for (int j = 0; j < TOPK; ++j) { lv[q + j] = NEG_INF; li[q + j] = 0; }
    float mv = NEG_INF; int mp = 0;
    const size_t base = (size_t)row * SLICES * TOPK;
    for (int j = 0; j < SLICES * TOPK; ++j) {
        const float v = pvals[base + j];
        if (v > mv) {
            lv[q + mp] = v; li[q + mp] = pidx[base + j];
            mv = lv[q]; mp = 0;
            #pragma unroll
            for (int t = 1; t < TOPK; ++t) {
                const float tv = lv[q + t];
                if (tv < mv) { mv = tv; mp = t; }
            }
        }
    }
    for (int j = 0; j < TOPK; ++j)
        topk_out[(size_t)row * TOPK + j] = li[q + j];
}

// ---------------------------------------------------------------------------
// x_hat[b] = sum_j codebook[topk[b][j]]  (fp32, un-normalized rows)
// ---------------------------------------------------------------------------
__global__ __launch_bounds__(256) void gather_sum(
    const float* __restrict__ codebook,
    const int* __restrict__ topk_idx,
    float* __restrict__ out)
{
    const int b = blockIdx.x;
    const int d = threadIdx.x * 4;
    float4 acc = {0.f, 0.f, 0.f, 0.f};
    for (int j = 0; j < TOPK; ++j) {
        const int idx = topk_idx[(size_t)b * TOPK + j];
        const float4 v = *(const float4*)&codebook[(size_t)idx * KDIM + d];
        acc.x += v.x; acc.y += v.y; acc.z += v.z; acc.w += v.w;
    }
    *(float4*)&out[(size_t)b * KDIM + d] = acc;
}

// ---------------------------------------------------------------------------
extern "C" void kernel_launch(void* const* d_in, const int* in_sizes, int n_in,
                              void* d_out, int out_size, void* d_ws, size_t ws_size,
                              hipStream_t stream) {
    (void)in_sizes; (void)n_in; (void)out_size; (void)ws_size;
    const float* x        = (const float*)d_in[0];   // [4096 x 1024] f32
    const float* codebook = (const float*)d_in[1];   // [16384 x 1024] f32
    float* out = (float*)d_out;

    unsigned char* ws = (unsigned char*)d_ws;
    size_t off = 0;
    unsigned short* xn = (unsigned short*)(ws + off); off += (size_t)B_ROWS * KDIM * 2;   // 8 MB
    unsigned short* cn = (unsigned short*)(ws + off); off += (size_t)N_CODES * KDIM * 2;  // 32 MB
    float* pvals = (float*)(ws + off); off += (size_t)B_ROWS * SLICES * TOPK * 4;         // 8 MB
    int*   pidx  = (int*)  (ws + off); off += (size_t)B_ROWS * SLICES * TOPK * 4;         // 8 MB
    int*   topk  = (int*)  (ws + off);                                                    // 0.5 MB

    normalize_bf16<<<B_ROWS,  256, 0, stream>>>(x, xn);
    normalize_bf16<<<N_CODES, 256, 0, stream>>>(codebook, cn);
    fused_cos_topk<<<dim3(SLICES, B_ROWS / ROWS), 256, SMEM_BYTES, stream>>>(xn, cn, pvals, pidx);
    merge_topk<<<B_ROWS / 128, 128, 0, stream>>>(pvals, pidx, topk);
    gather_sum<<<B_ROWS, 256, 0, stream>>>(codebook, topk, out);
}
